// Conv2dQInt8PerChannelModuleBase_81552839016704
// MI455X (gfx1250) — 8-bit matrix operand order repaired and checked by reading, compiled, not run on hardware
//
#include <hip/hip_runtime.h>
#include <stdint.h>

typedef int v8i __attribute__((ext_vector_type(8)));
typedef int v4i __attribute__((ext_vector_type(4)));

#define NB    32
#define CIN   256
#define COUT  256
#define H     56
#define W     56
#define OH    54
#define OW    54
#define KTOT  2304
#define NKSTEP 36

#define NHWC_BYTES  ((size_t)NB * H * W * CIN)
#define WPACK_OFF   NHWC_BYTES
#define WPACK_BYTES ((size_t)16 * NKSTEP * 32 * 32)
#define SUMW_OFF    (WPACK_OFF + WPACK_BYTES)

#define STRIP_B     (3 * W * 64)

#if defined(__has_builtin)
#if __has_builtin(__builtin_amdgcn_global_load_async_to_lds_b128)
#define HAVE_ASYNC_LDS 1
#endif
#endif

typedef __attribute__((address_space(1))) v4i as1_v4i;
typedef __attribute__((address_space(3))) v4i as3_v4i;

__device__ __forceinline__ void cp16_g2l(const void* g, void* l) {
#ifdef HAVE_ASYNC_LDS
    __builtin_amdgcn_global_load_async_to_lds_b128((as1_v4i*)g, (as3_v4i*)l, 0, 0);
#else
    *(v4i*)l = *(const v4i*)g;
#endif
}

__device__ __forceinline__ void wait_async() {
#ifdef HAVE_ASYNC_LDS
#if __has_builtin(__builtin_amdgcn_s_wait_asynccnt)
    __builtin_amdgcn_s_wait_asynccnt(0);
#else
    asm volatile("s_wait_asynccnt 0x0" ::: "memory");
#endif
#endif
}

__device__ __forceinline__ int dot4_ones(uint32_t u, int c) {
#if defined(__has_builtin)
#if __has_builtin(__builtin_amdgcn_sdot4)
    return __builtin_amdgcn_sdot4(u, 0x01010101u, c, false);
#else
    return c + (int)(int8_t)(u) + (int)(int8_t)(u >> 8)
             + (int)(int8_t)(u >> 16) + (int)(int8_t)(u >> 24);
#endif
#else
    return c + (int)(int8_t)(u) + (int)(int8_t)(u >> 8)
             + (int)(int8_t)(u >> 16) + (int)(int8_t)(u >> 24);
#endif
}

__global__ void k_nchw_to_nhwc(const int8_t* __restrict__ in, int8_t* __restrict__ out) {
    int t = blockIdx.x * blockDim.x + threadIdx.x;
    int p  = t >> 6;
    if (p >= NB * H * W) return;
    int c4 = (t & 63) * 4;
    int n = p / (H * W);
    int hw = p % (H * W);
    const uint8_t* base = (const uint8_t*)in + (size_t)n * CIN * H * W + hw;
    uint32_t v = 0;
#pragma unroll
    for (int i = 0; i < 4; ++i)
        v |= (uint32_t)base[(size_t)(c4 + i) * H * W] << (8 * i);
    *(uint32_t*)((uint8_t*)out + (size_t)p * CIN + c4) = v;
}

__global__ void k_pack_weights(const int8_t* __restrict__ w, int8_t* __restrict__ wp) {
    int t = blockIdx.x * blockDim.x + threadIdx.x;
    if (t >= 16 * NKSTEP * 32) return;
    int lane  = t & 31;
    int kstep = (t >> 5) % NKSTEP;
    int mtile = t / (32 * NKSTEP);
    int half = lane >> 4;
    int o = mtile * 16 + (lane & 15);
    int chunk = kstep / 9, rs = kstep % 9;
    alignas(16) uint8_t bytes[32];
#pragma unroll
    for (int b = 0; b < 32; ++b) {
        int vv = b >> 2, j = b & 3;
        int kk = ((vv >> 2) * 32) + half * 16 + ((vv & 3) * 4) + j;
        int c = chunk * 64 + kk;
        bytes[b] = (uint8_t)w[(size_t)o * KTOT + (size_t)c * 9 + rs];
    }
    v4i* dst = (v4i*)(wp + (((size_t)mtile * NKSTEP + kstep) * 32 + lane) * 32);
    dst[0] = *(const v4i*)&bytes[0];
    dst[1] = *(const v4i*)&bytes[16];
}

__global__ void k_weight_sums(const int8_t* __restrict__ w, int* __restrict__ sumw) {
    int o = threadIdx.x;
    int s = 0;
    for (int i = 0; i < KTOT; ++i) s += (int)w[(size_t)o * KTOT + i];
    sumw[o] = s;
}

__global__ __launch_bounds__(256, 1) void k_conv_wmma(
    const int8_t* __restrict__ xin,
    const int8_t* __restrict__ wpack,
    const int*    __restrict__ sumw,
    const float*  __restrict__ scales,
    const int*    __restrict__ zps,
    const float*  __restrict__ bias,
    int*          __restrict__ out)
{
    __shared__ uint8_t s_in[2][STRIP_B];
    __shared__ int s_colsum[64];

    const int bid = blockIdx.x;
    const int n = bid / OH, y = bid % OH;
    const int tid = threadIdx.x;
    const int wave = tid >> 5, lane = tid & 31;
    const int lane15 = lane & 15, half = lane >> 4;
    const int coutBase = wave * 32;

    if (tid < 64) s_colsum[tid] = 0;

    v8i acc[2][4];
#pragma unroll
    for (int i = 0; i < 2; ++i)
#pragma unroll
        for (int j = 0; j < 4; ++j) acc[i][j] = (v8i)0;

    const int8_t* xrow = xin + ((size_t)n * H + y) * W * CIN;

    for (int i = tid; i < 672; i += 256) {
        int seg = i >> 2, part = i & 3;
        int r = seg / W, x = seg % W;
        const int8_t* g = xrow + ((size_t)r * W + x) * CIN +   part * 16;
        cp16_g2l(g, s_in[0] + seg * 64 + part * 16);
    }
    wait_async();
    __syncthreads();

    for (int chunk = 0; chunk < 4; ++chunk) {
        const int buf = chunk & 1;
        const uint8_t* sb = s_in[buf];

        if (chunk < 3) {
            for (int i = tid; i < 672; i += 256) {
                int seg = i >> 2, part = i & 3;
                int r = seg / W, x = seg % W;
                const int8_t* g = xrow + ((size_t)r * W + x) * CIN
                                + (chunk + 1) * 64 + part * 16;
                cp16_g2l(g, s_in[buf ^ 1] + seg * 64 + part * 16);
            }
        }

        if (tid < W) {
            int s = 0;
#pragma unroll
            for (int r = 0; r < 3; ++r) {
                const uint32_t* p = (const uint32_t*)(sb + (r * W + tid) * 64);
#pragma unroll
                for (int q = 0; q < 16; ++q) s = dot4_ones(p[q], s);
            }
            s_colsum[tid] += s;
        }

#pragma unroll
        for (int r = 0; r < 3; ++r) {
#pragma unroll
            for (int s = 0; s < 3; ++s) {
                const int kstep = chunk * 9 + r * 3 + s;
                v8i a0 = *(const v8i*)(wpack +
                    (((size_t)(wave * 2 + 0) * NKSTEP + kstep) * 32 + lane) * 32);
                v8i a1 = *(const v8i*)(wpack +
                    (((size_t)(wave * 2 + 1) * NKSTEP + kstep) * 32 + lane) * 32);
#pragma unroll
                for (int nt = 0; nt < 4; ++nt) {
                    int xx = nt * 16 + lane15;
                    if (xx > OW - 1) xx = OW - 1;
                    int off = (r * W + (xx + s)) * 64 + half * 16;
                    v4i b0 = *(const v4i*)(sb + off);
                    v4i b1 = *(const v4i*)(sb + off + 32);
                    v8i bf;
                    bf[0] = b0[0]; bf[1] = b0[1]; bf[2] = b0[2]; bf[3] = b0[3];
                    bf[4] = b1[0]; bf[5] = b1[1]; bf[6] = b1[2]; bf[7] = b1[3];
                    acc[0][nt] = __builtin_amdgcn_wmma_i32_16x16x64_iu8(
                        true, a0, true, bf, acc[0][nt], false, false);
                    acc[1][nt] = __builtin_amdgcn_wmma_i32_16x16x64_iu8(
                        true, a1, true, bf, acc[1][nt], false, false);
                }
            }
        }

        wait_async();
        __syncthreads();
    }

#pragma unroll
    for (int mt = 0; mt < 2; ++mt) {
#pragma unroll
        for (int g = 0; g < 8; ++g) {
            const int o  = coutBase + mt * 16 + g + half * 8;
            const int zp = zps[o];
            const int sw = sumw[o];
            const float sc = scales[o];
            const float bo = bias[o];
            const int wconst = 7 * zp * KTOT - 7 * sw;
#pragma unroll
            for (int nt = 0; nt < 4; ++nt) {
                const int x = nt * 16 + lane15;
                if (x < OW) {
                    int sx = s_colsum[x] + s_colsum[x + 1] + s_colsum[x + 2];
                    int corr = acc[mt][nt][g] - zp * sx + wconst;
                    float v = 0.01f * sc * (float)corr + bo;
                    float rv = rintf(v);
                    int q;
                    if (rv >= 2147483647.0f)        q = 2147483647;
                    else if (rv <= -2147483648.0f)  q = (int)0x80000000;
                    else                            q = (int)rv;
                    out[(((size_t)n * COUT + o) * OH + y) * OW + x] = q;
                }
            }
        }
    }
}

extern "C" void kernel_launch(void* const* d_in, const int* in_sizes, int n_in,
                              void* d_out, int out_size, void* d_ws, size_t ws_size,
                              hipStream_t stream) {
    (void)in_sizes; (void)n_in; (void)out_size; (void)ws_size;
    const int8_t* xin    = (const int8_t*)d_in[0];
    const int8_t* wgt    = (const int8_t*)d_in[1];
    const float*  scales = (const float*)d_in[2];
    const int*    zps    = (const int*)d_in[3];
    const float*  bias   = (const float*)d_in[4];

    int8_t* nhwc  = (int8_t*)d_ws;
    int8_t* wpack = (int8_t*)d_ws + WPACK_OFF;
    int*    sumw  = (int*)((int8_t*)d_ws + SUMW_OFF);

    {
        int total = NB * H * W * 64;
        k_nchw_to_nhwc<<<(total + 255) / 256, 256, 0, stream>>>(xin, nhwc);
    }
    k_pack_weights<<<(16 * NKSTEP * 32 + 255) / 256, 256, 0, stream>>>(wgt, wpack);
    k_weight_sums<<<1, 256, 0, stream>>>(wgt, sumw);
    k_conv_wmma<<<NB * OH, 256, 0, stream>>>(nhwc, wpack, sumw, scales, zps, bias,
                                             (int*)d_out);
}
